// QuantizedLinear_58660663329081
// MI455X (gfx1250) — matrix operand order repaired and checked by reading, not compiled, not run on hardware
//
#include <hip/hip_runtime.h>

typedef __bf16 bf16_t;
typedef __attribute__((ext_vector_type(4)))  __bf16 v4bf;
typedef __attribute__((ext_vector_type(16))) __bf16 v16bf;
typedef __attribute__((ext_vector_type(8)))  float  v8f;

#define TOKENS 4096
#define IN_F   4096
#define OUT_F  11008

#define BM 128
#define BN 128
#define BK 32
#define LDSK (BK + 8)              // padded K stride (80 B rows: 16B-aligned, bank-staggered)
#define NKT (IN_F / BK)            // 128 k-steps
#define ABUF (BM * LDSK)           // elements per A buffer
#define BBUF (BN * LDSK)           // elements per B buffer

// ---------------- pass 1a: rowsum(x), one block per token ----------------
__global__ __launch_bounds__(256) void rowsum_kernel(const float* __restrict__ x,
                                                     float* __restrict__ rs) {
    __shared__ float red[256];
    const float* row = x + (size_t)blockIdx.x * IN_F;
    float s = 0.f;
    for (int i = threadIdx.x; i < IN_F; i += 256) s += row[i];
    red[threadIdx.x] = s;
    __syncthreads();
    for (int off = 128; off > 0; off >>= 1) {
        if ((int)threadIdx.x < off) red[threadIdx.x] += red[threadIdx.x + off];
        __syncthreads();
    }
    if (threadIdx.x == 0) rs[blockIdx.x] = red[0];
}

// ---------------- pass 1b: Wq int32 -> bf16 (exact: values 0..255) ----------------
__global__ __launch_bounds__(256) void convert_w_kernel(const int* __restrict__ wq,
                                                        bf16_t* __restrict__ wb) {
    size_t idx = ((size_t)blockIdx.x * 256 + threadIdx.x) * 4;
    int4 v = *(const int4*)(wq + idx);
    v4bf p = { (bf16_t)(float)v.x, (bf16_t)(float)v.y,
               (bf16_t)(float)v.z, (bf16_t)(float)v.w };
    *(v4bf*)(wb + idx) = p;
}

// ---------------- pass 1c: x f32 -> bf16 ----------------
__global__ __launch_bounds__(256) void convert_x_kernel(const float* __restrict__ x,
                                                        bf16_t* __restrict__ xb) {
    size_t idx = ((size_t)blockIdx.x * 256 + threadIdx.x) * 4;
    float4 v = *(const float4*)(x + idx);
    v4bf p = { (bf16_t)v.x, (bf16_t)v.y, (bf16_t)v.z, (bf16_t)v.w };
    *(v4bf*)(xb + idx) = p;
}

// ---------------- pass 2: bf16 WMMA GEMM, 128 threads (4 waves, 1/SIMD) ----------------
// Copies use GLOBAL_LOAD_ASYNC_TO_LDS_B128 (GVS mode): zero staging VGPRs,
// ASYNCcnt-tracked background DMA into the double-buffered LDS tiles.
__global__ __launch_bounds__(128, 1) void qlinear_wmma_kernel(
    const bf16_t* __restrict__ xb,     // [TOKENS, IN_F] bf16
    const bf16_t* __restrict__ wb,     // [OUT_F, IN_F]  bf16
    const float*  __restrict__ scale,  // [OUT_F]
    const float*  __restrict__ zero,   // [OUT_F]
    const float*  __restrict__ bias,   // [OUT_F]
    const float*  __restrict__ rowsum, // [TOKENS]
    float*        __restrict__ out)    // [TOKENS, OUT_F]
{
    __shared__ bf16_t As[2 * ABUF];
    __shared__ bf16_t Bs[2 * BBUF];

    const int tid    = threadIdx.x;
    const int lane   = tid & 31;
    const int waveId = tid >> 5;      // 0..3
    const int lane16 = lane & 15;
    const int khalf  = lane >> 4;

    const int wm  = waveId & 1;       // 2 wave-rows of 64
    const int wnc = waveId >> 1;      // 2 wave-cols of 64

    const int rowBase = blockIdx.y * BM;
    const int colBase = blockIdx.x * BN;

    // ---- async-copy addressing: thread tid owns row tid of each 128x32 tile ----
    // global: SGPR base + 32-bit VGPR byte offset (GVS); k-advance = +64B per step.
    // LDS: VGPR byte offset; the shared 24-bit inst offset (0/16/32/48) applies to
    // BOTH sides, which matches a contiguous 64B row copy on both sides.
    unsigned aGOff = (unsigned)((size_t)(rowBase + tid) * IN_F * sizeof(bf16_t));
    unsigned bGOff = (unsigned)((size_t)(colBase + tid) * IN_F * sizeof(bf16_t));
    const unsigned aLds0 = (unsigned)(uintptr_t)&As[tid * LDSK];
    const unsigned aLds1 = (unsigned)(uintptr_t)&As[ABUF + tid * LDSK];
    const unsigned bLds0 = (unsigned)(uintptr_t)&Bs[tid * LDSK];
    const unsigned bLds1 = (unsigned)(uintptr_t)&Bs[BBUF + tid * LDSK];

    // fragment row bases (elements)
    const int aFragRow = (wm  * 64 + lane16) * LDSK;   // + mi*16*LDSK
    const int bFragRow = (wnc * 64 + lane16) * LDSK;   // + ni*16*LDSK

    v8f acc[4][4];
    const v8f zv = {0.f,0.f,0.f,0.f,0.f,0.f,0.f,0.f};
    #pragma unroll
    for (int i = 0; i < 4; ++i)
        #pragma unroll
        for (int j = 0; j < 4; ++j) acc[i][j] = zv;

    // issue 8 async row-copies (4 A + 4 B) for one k-step into buffer `buf`
    auto asyncCopy = [&](int buf) {
        const unsigned al = buf ? aLds1 : aLds0;
        const unsigned bl = buf ? bLds1 : bLds0;
        asm volatile("global_load_async_to_lds_b128 %0, %1, %2 offset:0"
                     :: "v"(al), "v"(aGOff), "s"(xb) : "memory");
        asm volatile("global_load_async_to_lds_b128 %0, %1, %2 offset:16"
                     :: "v"(al), "v"(aGOff), "s"(xb) : "memory");
        asm volatile("global_load_async_to_lds_b128 %0, %1, %2 offset:32"
                     :: "v"(al), "v"(aGOff), "s"(xb) : "memory");
        asm volatile("global_load_async_to_lds_b128 %0, %1, %2 offset:48"
                     :: "v"(al), "v"(aGOff), "s"(xb) : "memory");
        asm volatile("global_load_async_to_lds_b128 %0, %1, %2 offset:0"
                     :: "v"(bl), "v"(bGOff), "s"(wb) : "memory");
        asm volatile("global_load_async_to_lds_b128 %0, %1, %2 offset:16"
                     :: "v"(bl), "v"(bGOff), "s"(wb) : "memory");
        asm volatile("global_load_async_to_lds_b128 %0, %1, %2 offset:32"
                     :: "v"(bl), "v"(bGOff), "s"(wb) : "memory");
        asm volatile("global_load_async_to_lds_b128 %0, %1, %2 offset:48"
                     :: "v"(bl), "v"(bGOff), "s"(wb) : "memory");
        aGOff += BK * sizeof(bf16_t);
        bGOff += BK * sizeof(bf16_t);
    };
    auto waitAsync = [&]() {
        asm volatile("s_wait_asynccnt 0x0" ::: "memory");
    };

    // A fragment: lane(0-15)=row M, element e -> K = e + 16*khalf
    auto loadA = [&](int buf, int mi) -> v16bf {
        v16bf f;
        const bf16_t* base = &As[(buf ? ABUF : 0) + aFragRow + mi * 16 * LDSK];
        ((uint4*)&f)[0] = *(const uint4*)(base + 16 * khalf);       // K 0-7 / 16-23
        ((uint4*)&f)[1] = *(const uint4*)(base + 16 * khalf + 8);   // K 8-15 / 24-31
        return f;
    };
    // B fragment: lane16 = column N, khalf selects K half, elems contiguous
    auto loadB = [&](int buf, int ni) -> v16bf {
        v16bf f;
        const bf16_t* base = &Bs[(buf ? BBUF : 0) + bFragRow + ni * 16 * LDSK + 16 * khalf];
        ((uint4*)&f)[0] = ((const uint4*)base)[0];
        ((uint4*)&f)[1] = ((const uint4*)base)[1];
        return f;
    };
    auto computeTile = [&](int buf) {
        v16bf af[4], bq[4];
        #pragma unroll
        for (int i = 0; i < 4; ++i) af[i] = loadA(buf, i);
        #pragma unroll
        for (int i = 0; i < 4; ++i) bq[i] = loadB(buf, i);
        #pragma unroll
        for (int ni = 0; ni < 4; ++ni)
            #pragma unroll
            for (int mi = 0; mi < 4; ++mi)
                acc[mi][ni] = __builtin_amdgcn_wmma_f32_16x16x32_bf16(
                    false, af[mi], false, bq[ni], (short)0, acc[mi][ni], false, false);
    };

    // pipelined main loop: asyncs for k+1 run in background during compute of k
    asyncCopy(0);
    waitAsync();
    __syncthreads();
    for (int kt = 0; kt < NKT; ++kt) {
        if (kt + 1 < NKT) asyncCopy((kt + 1) & 1);   // other buffer: no race
        computeTile(kt & 1);
        if (kt + 1 < NKT) {
            waitAsync();          // this wave's async copies landed in LDS
            __syncthreads();      // all waves' copies landed
        }
    }

    // epilogue: y = s*acc - s*z*rowsum + b   (C/D layout: VGPR i -> M = i + 8*khalf)
    #pragma unroll
    for (int mi = 0; mi < 4; ++mi) {
        const int rbase = rowBase + wm * 64 + mi * 16 + 8 * khalf;
        #pragma unroll
        for (int ni = 0; ni < 4; ++ni) {
            const int col = colBase + wnc * 64 + ni * 16 + lane16;
            const float s  = scale[col];
            const float sz = s * zero[col];
            const float b  = bias[col];
            #pragma unroll
            for (int vi = 0; vi < 8; ++vi) {
                const int row = rbase + vi;
                out[(size_t)row * OUT_F + col] =
                    acc[mi][ni][vi] * s - sz * rowsum[row] + b;
            }
        }
    }
}

extern "C" void kernel_launch(void* const* d_in, const int* in_sizes, int n_in,
                              void* d_out, int out_size, void* d_ws, size_t ws_size,
                              hipStream_t stream) {
    const float* x     = (const float*)d_in[0];
    const int*   wq    = (const int*)  d_in[1];
    const float* scale = (const float*)d_in[2];
    const float* zero  = (const float*)d_in[3];
    const float* bias  = (const float*)d_in[4];
    float* out = (float*)d_out;

    // workspace layout (all 16B aligned):
    //   [0, 16KB)            rowsum (TOKENS f32)
    //   [16KB, +32MB)        xb     (TOKENS*IN_F bf16)
    //   [16KB+32MB, +90MB)   wb     (OUT_F*IN_F bf16)
    char* ws = (char*)d_ws;
    float*  rowsum = (float*)ws;
    bf16_t* xb     = (bf16_t*)(ws + 16384);
    bf16_t* wb     = (bf16_t*)(ws + 16384 + (size_t)TOKENS * IN_F * sizeof(bf16_t));

    rowsum_kernel<<<TOKENS, 256, 0, stream>>>(x, rowsum);
    convert_x_kernel<<<(TOKENS * (size_t)IN_F) / 1024, 256, 0, stream>>>(x, xb);
    convert_w_kernel<<<((size_t)OUT_F * IN_F) / 1024, 256, 0, stream>>>(wq, wb);

    dim3 grid(OUT_F / BN, TOKENS / BM);   // 86 x 32 blocks, 128 threads each
    qlinear_wmma_kernel<<<grid, 128, 0, stream>>>(xb, wb, scale, zero, bias, rowsum, out);
}
